// Translator_90666759619093
// MI455X (gfx1250) — matrix operand order repaired and checked by reading, not compiled, not run on hardware
//
#include <hip/hip_runtime.h>
#include <hip/hip_bf16.h>
#include <math.h>

// ---------------------------------------------------------------------------
// Problem constants (from reference)
// ---------------------------------------------------------------------------
#define BEAM    4
#define MAXLEN  48
#define VOCAB   32000
#define DMODEL  512
#define BOS     2
#define EOS     3
#define PAD     1
#define BATCH   16
#define SRCLEN  64
#define NEGV    (-1.0e20f)
#define NROWS   (BATCH * BEAM)          // 64
#define NSTEPS  (MAXLEN - 2)            // 46
#define COLT    (VOCAB / 16)            // 2000 column tiles of 16
#define NBLK    (COLT / 8)              // 250 GEMM workgroups, 128 cols each

typedef __attribute__((ext_vector_type(16))) __bf16 v16bf;
typedef __attribute__((ext_vector_type(8)))  float  v8f;

// ---------------------------------------------------------------------------
// top-4 online insert (descending, stable on ties because of strict >)
// ---------------------------------------------------------------------------
__device__ __forceinline__ void top4_insert(float v, int idx, float* tv, int* ti) {
    if (v > tv[3]) {
        int p = 3;
        while (p > 0 && v > tv[p - 1]) { tv[p] = tv[p - 1]; ti[p] = ti[p - 1]; --p; }
        tv[p] = v; ti[p] = idx;
    }
}

// ---------------------------------------------------------------------------
// Kernel 0: convert W [512 x 32000] f32 -> bf16, swizzled into the per-lane
// layout a WMMA B operand (32x16, bf16) wants:
//   vector index = (nt*16 + kc)*32 + lane ; element e -> K = kc*32 + (lane/16)*8 + (e&7) + (e>>3)*16,
//   N = nt*16 + (lane%16).
// ---------------------------------------------------------------------------
__global__ __launch_bounds__(256) void k_wswizzle(const float* __restrict__ W,
                                                  __bf16* __restrict__ Wt) {
    int id = blockIdx.x * 256 + threadIdx.x;          // (nt,kc,lane)
    if (id >= COLT * 16 * 32) return;
    int lane = id & 31;
    int kc   = (id >> 5) & 15;
    int nt   = id >> 9;
    int n    = nt * 16 + (lane & 15);
    int kb   = kc * 32 + (lane >> 4) * 8;
    __bf16* dst = Wt + (size_t)id * 16;
#pragma unroll
    for (int e = 0; e < 16; ++e)
        dst[e] = (__bf16)W[(size_t)(kb + (e & 7) + ((e >> 3) << 4)) * VOCAB + n];
}

// ---------------------------------------------------------------------------
// ctx[b][d] = mean over SRCLEN of enc_out[b][s][d]
// ---------------------------------------------------------------------------
__global__ __launch_bounds__(256) void k_ctx(const float* __restrict__ enc,
                                             float* __restrict__ ctx) {
    int id = blockIdx.x * 256 + threadIdx.x;          // 16*512
    if (id >= BATCH * DMODEL) return;
    int b = id >> 9, d = id & (DMODEL - 1);
    float s = 0.f;
    for (int t = 0; t < SRCLEN; ++t)
        s += enc[((size_t)b * SRCLEN + t) * DMODEL + d];
    ctx[id] = s * (1.0f / SRCLEN);
}

// ---------------------------------------------------------------------------
// init beam-search state
// ---------------------------------------------------------------------------
__global__ __launch_bounds__(256) void k_init(int* __restrict__ hyp0,
                                              float* __restrict__ scores,
                                              float* __restrict__ finScore,
                                              int* __restrict__ finSeq,
                                              float* __restrict__ lp) {
    int t = threadIdx.x;
    for (int i = t; i < NROWS * MAXLEN; i += 256) {
        int row = i / MAXLEN, p = i % MAXLEN;
        hyp0[i] = (p == 0 && (row & 3) == 0) ? BOS : PAD;
    }
    for (int i = t; i < NROWS; i += 256)             // scores [16][4]
        scores[i] = ((i & 3) == 0) ? 0.0f : NEGV;
    for (int i = t; i < BATCH; i += 256) finScore[i] = NEGV;
    for (int i = t; i < BATCH * MAXLEN; i += 256) finSeq[i] = PAD;
    for (int i = t; i < MAXLEN; i += 256)
        lp[i] = powf(6.0f / (5.0f + (float)i), 0.6f);
}

// ---------------------------------------------------------------------------
// Kernel A: dec = tanh(embed[cur] + ctx) -> bf16, swizzled into the WMMA
// A-operand layout (16x32 bf16 per ISA table):
//   lane = ((k%32>>3)&1)*16 + row%16 ; e = (k%32 & 7) | ((k%32>>4)<<3)
// ---------------------------------------------------------------------------
__global__ __launch_bounds__(128) void k_dec(const float* __restrict__ embed,
                                             const float* __restrict__ ctx,
                                             const int* __restrict__ hyp_cur,
                                             __bf16* __restrict__ dec_sw,
                                             int step) {
    int row = blockIdx.x;                             // 64
    int tok = hyp_cur[row * MAXLEN + step];
    int mt = row >> 4, l = row & 15;
    const float* e = embed + (size_t)tok * DMODEL;
    const float* c = ctx + (size_t)(row >> 2) * DMODEL;
    for (int d = threadIdx.x; d < DMODEL; d += 128) {
        float v = tanhf(e[d] + c[d]);
        int kc  = d >> 5, k32 = d & 31;
        int h   = (k32 >> 3) & 1;
        int el  = (k32 & 7) | ((k32 >> 4) << 3);
        int lane = h * 16 + l;
        dec_sw[(((mt * 16 + kc) * 32 + lane) << 4) + el] = (__bf16)v;
    }
}

// ---------------------------------------------------------------------------
// Kernel B: bf16 WMMA GEMM (64 x 128 tile per workgroup, K=512) fused with
// per-row online max / sum-exp / top-4 over this block's 128 columns.
//
// CDNA5 structure:
//  - A panel (64 KB, shared by all 8 waves) is staged into LDS once with
//    global_load_async_to_lds_b128 (ASYNCcnt), then served by ds_load_b128
//    (DScnt) so inner-loop WMMAs never wait on global latency for A.
//  - B tile is double-buffered in registers (LOADcnt), with a global_prefetch
//    one chunk ahead, so the 4 WMMAs of chunk kc overlap the B load of kc+1.
//  - The same 64 KB LDS buffer is reused (after a barrier) to stage z-tiles
//    for the per-row online softmax/top-4 reduction.
// ---------------------------------------------------------------------------
__global__ __launch_bounds__(256) void k_gemm(const __bf16* __restrict__ dec_sw,
                                              const __bf16* __restrict__ Wt_sw,
                                              float* __restrict__ pM,
                                              float* __restrict__ pS,
                                              float* __restrict__ pV,
                                              int*   __restrict__ pI) {
    __shared__ char smem[NROWS * DMODEL * 2];         // 64 KB, dual-purpose
    int tid  = threadIdx.x;
    int wave = tid >> 5;
    int lane = tid & 31;
    int nt   = blockIdx.x * 8 + wave;                 // column tile (16 cols)

    // ---- stage the 64 KB A panel into LDS via async copy (ASYNCcnt) ----
    {
        const char* g = (const char*)dec_sw;
#pragma unroll
        for (int q = 0; q < 16; ++q) {
            int byteoff = (q * 256 + tid) * 16;       // 4096 x b128 chunks
            unsigned lds_off = (unsigned)(uintptr_t)(smem + byteoff);
            const char* gp = g + byteoff;
            asm volatile("global_load_async_to_lds_b128 %0, %1, off"
                         :: "v"(lds_off), "v"(gp) : "memory");
        }
        asm volatile("s_wait_asynccnt 0x0" ::: "memory");
    }
    __syncthreads();

    const v16bf* AL = (const v16bf*)smem;             // LDS A panel
    const v16bf* Bp = (const v16bf*)Wt_sw + (size_t)nt * 16 * 32 + lane;

    v8f acc0 = {}; v8f acc1 = {}; v8f acc2 = {}; v8f acc3 = {};
    v16bf bcur = Bp[0];
#pragma unroll
    for (int kc = 0; kc < 16; ++kc) {
        v16bf bnxt = bcur;
        if (kc < 15) bnxt = Bp[(kc + 1) * 32];        // double-buffered B
        if (kc < 14)
            __builtin_prefetch((const void*)&Bp[(kc + 2) * 32], 0, 3);
        v16bf a0 = AL[(size_t)(0 * 16 + kc) * 32 + lane];   // ds_load_b128 x2
        v16bf a1 = AL[(size_t)(1 * 16 + kc) * 32 + lane];
        v16bf a2 = AL[(size_t)(2 * 16 + kc) * 32 + lane];
        v16bf a3 = AL[(size_t)(3 * 16 + kc) * 32 + lane];
        acc0 = __builtin_amdgcn_wmma_f32_16x16x32_bf16(false, a0, false, bcur, (short)0, acc0, false, false);
        acc1 = __builtin_amdgcn_wmma_f32_16x16x32_bf16(false, a1, false, bcur, (short)0, acc1, false, false);
        acc2 = __builtin_amdgcn_wmma_f32_16x16x32_bf16(false, a2, false, bcur, (short)0, acc2, false, false);
        acc3 = __builtin_amdgcn_wmma_f32_16x16x32_bf16(false, a3, false, bcur, (short)0, acc3, false, false);
        bcur = bnxt;
    }
    __syncthreads();                                  // done reading A panel

    // ---- reuse LDS as z-tile staging: zs[wave][row][col] ----
    float* zs = (float*)smem;                         // 8*64*16 floats = 32 KB
    int roff = (lane >> 4) * 8;
    int col  = lane & 15;
#pragma unroll
    for (int r = 0; r < 8; ++r) {
        zs[(wave * 64 + ( 0 + roff + r)) * 16 + col] = acc0[r];
        zs[(wave * 64 + (16 + roff + r)) * 16 + col] = acc1[r];
        zs[(wave * 64 + (32 + roff + r)) * 16 + col] = acc2[r];
        zs[(wave * 64 + (48 + roff + r)) * 16 + col] = acc3[r];
    }
    __syncthreads();

    if (tid < 64) {
        int row = tid;
        float m = -3.0e38f, s = 0.0f;
        float tv[4] = {-3.0e38f, -3.0e38f, -3.0e38f, -3.0e38f};
        int   ti[4] = {0, 0, 0, 0};
        int nb0 = blockIdx.x * 128;
        for (int w = 0; w < 8; ++w)
            for (int c = 0; c < 16; ++c) {
                float z = zs[(w * 64 + row) * 16 + c];
                if (z > m) { s = s * __expf(m - z) + 1.0f; m = z; }
                else       { s += __expf(z - m); }
                top4_insert(z, nb0 + w * 16 + c, tv, ti);
            }
        int o = blockIdx.x * 64 + row;
        pM[o] = m; pS[o] = s;
#pragma unroll
        for (int j = 0; j < 4; ++j) { pV[o * 4 + j] = tv[j]; pI[o * 4 + j] = ti[j]; }
    }
}

// ---------------------------------------------------------------------------
// Kernel C: reduce the 250 block partials per row -> global m, s, top-4
// ---------------------------------------------------------------------------
__global__ __launch_bounds__(64) void k_reduce(const float* __restrict__ pM,
                                               const float* __restrict__ pS,
                                               const float* __restrict__ pV,
                                               const int*   __restrict__ pI,
                                               float* __restrict__ rM,
                                               float* __restrict__ rS,
                                               float* __restrict__ rV,
                                               int*   __restrict__ rI) {
    int row = threadIdx.x;
    if (row >= NROWS) return;
    float m = -3.0e38f, s = 0.0f;
    float tv[4] = {-3.0e38f, -3.0e38f, -3.0e38f, -3.0e38f};
    int   ti[4] = {0, 0, 0, 0};
    for (int b = 0; b < NBLK; ++b) {
        int o = b * 64 + row;
        float mb = pM[o], sb = pS[o];
        if (mb > m) { s = s * __expf(m - mb) + sb; m = mb; }
        else        { s += sb * __expf(mb - m); }
#pragma unroll
        for (int j = 0; j < 4; ++j)
            top4_insert(pV[o * 4 + j], pI[o * 4 + j], tv, ti);
    }
    rM[row] = m; rS[row] = s;
#pragma unroll
    for (int j = 0; j < 4; ++j) { rV[row * 4 + j] = tv[j]; rI[row * 4 + j] = ti[j]; }
}

// ---------------------------------------------------------------------------
// Kernel D: one beam-search step (exact reference semantics, incl. the
// length-penalized finished-candidate bookkeeping). Double-buffered hyp.
// ---------------------------------------------------------------------------
__global__ __launch_bounds__(256) void k_step(const float* __restrict__ rM,
                                              const float* __restrict__ rS,
                                              const float* __restrict__ rV,
                                              const int*   __restrict__ rI,
                                              const int*   __restrict__ hyp_old,
                                              int*   __restrict__ hyp_new,
                                              float* __restrict__ scores,
                                              float* __restrict__ finScore,
                                              int*   __restrict__ finSeq,
                                              const float* __restrict__ lp,
                                              int step) {
    __shared__ int   aOrg[BATCH][BEAM];   // global source row for new beam j
    __shared__ int   aTok[BATCH][BEAM];
    __shared__ int   fDo[BATCH], fRow[BATCH], fTok[BATCH];
    int t = threadIdx.x;

    if (t < BATCH) {
        int b = t;
        float cv[16]; int ctok[16], corg[16];
        for (int r4 = 0; r4 < BEAM; ++r4) {
            int row = b * BEAM + r4;
            float mm = rM[row];
            float ls = logf(rS[row]);
            float sc = scores[row];
            for (int j = 0; j < 4; ++j) {
                cv[r4 * 4 + j]   = sc + (rV[row * 4 + j] - mm - ls);
                ctok[r4 * 4 + j] = rI[row * 4 + j];
                corg[r4 * 4 + j] = r4;
            }
        }
        // global top-4 of 16 candidates (desc, stable)
        float topv[4]; int toptok[4], toporg[4];
        bool used[16];
        for (int q = 0; q < 16; ++q) used[q] = false;
        for (int k = 0; k < 4; ++k) {
            float best = -3.0e38f; int bi = 0;
            for (int q = 0; q < 16; ++q)
                if (!used[q] && cv[q] > best) { best = cv[q]; bi = q; }
            used[bi] = true;
            topv[k] = cv[bi]; toptok[k] = ctok[bi]; toporg[k] = corg[bi];
        }
        // alive masking + sort (desc, stable)
        float av[4];
        for (int j = 0; j < 4; ++j)
            av[j] = topv[j] + ((toptok[j] == EOS) ? NEGV : 0.0f);
        bool u2[4] = {false, false, false, false};
        for (int k = 0; k < 4; ++k) {
            float best = -3.0e38f; int bi = 0;
            for (int j = 0; j < 4; ++j)
                if (!u2[j] && av[j] > best) { best = av[j]; bi = j; }
            u2[bi] = true;
            scores[b * BEAM + k] = av[bi];
            aOrg[b][k] = b * BEAM + toporg[bi];
            aTok[b][k] = toptok[bi];
        }
        // finished bookkeeping (mirrors reference, including masked quirk)
        float fbest = -3.0e38f; int fidx = 0;
        for (int j = 0; j < 4; ++j) {
            float f = topv[j] + ((toptok[j] == EOS) ? 0.0f : NEGV);
            if (f > fbest) { fbest = f; fidx = j; }
        }
        float pen = fbest * lp[step + 1];
        if (pen > finScore[b]) {
            finScore[b] = pen;
            fDo[b] = 1; fRow[b] = b * BEAM + toporg[fidx]; fTok[b] = toptok[fidx];
        } else {
            fDo[b] = 0; fRow[b] = 0; fTok[b] = 0;
        }
    }
    __syncthreads();

    // new_hyp = cand[alive]: copy from hyp_old with token written at step+1
    for (int idx = t; idx < NROWS * MAXLEN; idx += 256) {
        int o = idx / MAXLEN, p = idx % MAXLEN;
        int b = o >> 2, j = o & 3;
        int src = aOrg[b][j];
        hyp_new[o * MAXLEN + p] =
            (p == step + 1) ? aTok[b][j] : hyp_old[src * MAXLEN + p];
    }
    // conditional finished-sequence update
    for (int idx = t; idx < BATCH * MAXLEN; idx += 256) {
        int b = idx / MAXLEN, p = idx % MAXLEN;
        if (fDo[b])
            finSeq[b * MAXLEN + p] =
                (p == step + 1) ? fTok[b] : hyp_old[fRow[b] * MAXLEN + p];
    }
}

// ---------------------------------------------------------------------------
// Kernel E: emit result [16][48] (tokens as floats) + final_scores [16]
// ---------------------------------------------------------------------------
__global__ __launch_bounds__(256) void k_final(const int* __restrict__ hyp_fin,
                                               const float* __restrict__ scores,
                                               const float* __restrict__ finScore,
                                               const int* __restrict__ finSeq,
                                               const float* __restrict__ lp,
                                               float* __restrict__ out) {
    int t = threadIdx.x;
    for (int idx = t; idx < BATCH * MAXLEN; idx += 256) {
        int b = idx / MAXLEN, p = idx % MAXLEN;
        bool hf = finScore[b] > NEGV * 0.5f;
        int v;
        if (p < MAXLEN - 1)
            v = hf ? finSeq[b * MAXLEN + p + 1]
                   : hyp_fin[(b * BEAM) * MAXLEN + p + 1];
        else
            v = PAD;
        out[idx] = (float)v;
    }
    if (t < BATCH) {
        bool hf = finScore[t] > NEGV * 0.5f;
        out[BATCH * MAXLEN + t] =
            hf ? finScore[t] : scores[t * BEAM + 0] * lp[MAXLEN - 1];
    }
}

// ---------------------------------------------------------------------------
// Launch: setup (W swizzle, ctx, init) + 46 x (dec, gemm, reduce, step) + final
// ---------------------------------------------------------------------------
extern "C" void kernel_launch(void* const* d_in, const int* in_sizes, int n_in,
                              void* d_out, int out_size, void* d_ws, size_t ws_size,
                              hipStream_t stream) {
    const float* enc   = (const float*)d_in[0];   // [16,64,512]
    const float* embed = (const float*)d_in[1];   // [32000,512]
    const float* W     = (const float*)d_in[2];   // [512,32000]
    float* out = (float*)d_out;

    char* base = (char*)d_ws;
    size_t off = 0;
    auto carve = [&](size_t bytes) -> char* {
        char* p = base + off;
        off = (off + bytes + 255) & ~(size_t)255;
        return p;
    };
    __bf16* Wt     = (__bf16*)carve((size_t)VOCAB * DMODEL * 2);  // 32.77 MB
    __bf16* dec_sw = (__bf16*)carve((size_t)NROWS * DMODEL * 2);  // 64 KB
    float*  ctx    = (float*) carve((size_t)BATCH * DMODEL * 4);
    int*    hyp    = (int*)   carve((size_t)2 * NROWS * MAXLEN * 4);
    float*  scores = (float*) carve((size_t)NROWS * 4);
    float*  finSc  = (float*) carve((size_t)BATCH * 4);
    int*    finSeq = (int*)   carve((size_t)BATCH * MAXLEN * 4);
    float*  lp     = (float*) carve((size_t)MAXLEN * 4);
    float*  pM     = (float*) carve((size_t)NBLK * NROWS * 4);
    float*  pS     = (float*) carve((size_t)NBLK * NROWS * 4);
    float*  pV     = (float*) carve((size_t)NBLK * NROWS * 4 * 4);
    int*    pI     = (int*)   carve((size_t)NBLK * NROWS * 4 * 4);
    float*  rM     = (float*) carve((size_t)NROWS * 4);
    float*  rS     = (float*) carve((size_t)NROWS * 4);
    float*  rV     = (float*) carve((size_t)NROWS * 4 * 4);
    int*    rI     = (int*)   carve((size_t)NROWS * 4 * 4);
    (void)in_sizes; (void)n_in; (void)out_size; (void)ws_size;

    // one-time (per launch) setup
    k_wswizzle<<<(COLT * 16 * 32 + 255) / 256, 256, 0, stream>>>(W, Wt);
    k_ctx<<<(BATCH * DMODEL + 255) / 256, 256, 0, stream>>>(enc, ctx);
    k_init<<<1, 256, 0, stream>>>(hyp, scores, finSc, finSeq, lp);

    for (int i = 0; i < NSTEPS; ++i) {
        int* hyp_cur = hyp + (size_t)(i & 1)        * NROWS * MAXLEN;
        int* hyp_nxt = hyp + (size_t)((i + 1) & 1)  * NROWS * MAXLEN;
        k_dec<<<NROWS, 128, 0, stream>>>(embed, ctx, hyp_cur, dec_sw, i);
        k_gemm<<<NBLK, 256, 0, stream>>>(dec_sw, Wt, pM, pS, pV, pI);
        k_reduce<<<1, 64, 0, stream>>>(pM, pS, pV, pI, rM, rS, rV, rI);
        k_step<<<1, 256, 0, stream>>>(rM, rS, rV, rI, hyp_cur, hyp_nxt,
                                      scores, finSc, finSeq, lp, i);
    }
    // after 46 steps, alive hypotheses live in buffer parity 0
    k_final<<<1, 256, 0, stream>>>(hyp, scores, finSc, finSeq, lp, out);
}
